// FlaxBasicTransformerBlock_6880537608432
// MI455X (gfx1250) — compile-verified, matrix operand order checked by reading, not run on hardware
//
#include <hip/hip_runtime.h>
#include <cstdint>
#include <cstddef>

// ---------------- problem constants ----------------
#define DIMD   1024
#define HEADS  16
#define DHEAD  64
#define FFD    4096
#define CTXD   768
#define NB     4
#define SEQ    2048
#define NTOK   (NB*SEQ)     // 8192
#define TCTX   77
#define TPADC  96
#define CROWS  (NB*TPADC)   // 384
#define ATT_SCALE 0.125f    // 64^-0.5

typedef _Float16 v8h  __attribute__((ext_vector_type(8)));
typedef _Float16 v16h __attribute__((ext_vector_type(16)));
typedef float    v8f  __attribute__((ext_vector_type(8)));
typedef int      v4i  __attribute__((vector_size(16)));

// ---- optional CDNA5 async global->LDS path (guarded; sync fallback) ----
#if defined(__has_builtin)
#if __has_builtin(__builtin_amdgcn_global_load_async_to_lds_b128) && \
    __has_builtin(__builtin_amdgcn_s_wait_asynccnt)
#define USE_ASYNC_LDS 1
#endif
#endif

typedef __attribute__((address_space(1))) v4i* as1_v4i_p;
typedef __attribute__((address_space(3))) v4i* as3_v4i_p;

// copy 16 bytes global -> LDS (per lane)
__device__ __forceinline__ void copy16_to_lds(const _Float16* g, _Float16* l) {
#ifdef USE_ASYNC_LDS
  __builtin_amdgcn_global_load_async_to_lds_b128(
      (as1_v4i_p)(void*)(const_cast<_Float16*>(g)),
      (as3_v4i_p)(void*)l, 0, 0);
#else
  *reinterpret_cast<v8h*>(l) = *reinterpret_cast<const v8h*>(g);
#endif
}

template <int N>
__device__ __forceinline__ void async_wait() {
#ifdef USE_ASYNC_LDS
  __builtin_amdgcn_s_wait_asynccnt(N);
#endif
}

// Load one 16x16x32 f16 WMMA A/B fragment for this lane.
// p must already include:  row_base + (lane>>4)*8   (row stride in elements).
// ISA layout: elements {0..7} at k = (lane>>4)*8, elements {8..15} at +16.
__device__ __forceinline__ v16h frag_load(const _Float16* p) {
  v8h lo = *reinterpret_cast<const v8h*>(p);
  v8h hi = *reinterpret_cast<const v8h*>(p + 16);
  return __builtin_shufflevector(lo, hi, 0,1,2,3,4,5,6,7,8,9,10,11,12,13,14,15);
}

__device__ __forceinline__ v8f wmma16(v16h a, v16h b, v8f c) {
  // D = A(16x32 f16) * B(32x16 f16) + C(16x16 f32)
  return __builtin_amdgcn_wmma_f32_16x16x32_f16(false, a, false, b, (short)0, c,
                                                false, false);
}

__device__ __forceinline__ float gelu_tanh(float x) {
  return 0.5f * x * (1.0f + tanhf(0.7978845608028654f * (x + 0.044715f * x * x * x)));
}

// ---------------- elementwise helpers ----------------

// transpose+cast weight: src [Kin, Nout] f32 row-major -> dst [Nout, Kin] f16
__global__ void wcast_tr(const float* __restrict__ src, _Float16* __restrict__ dst,
                         int Kin, int Nout) {
  long idx = (long)blockIdx.x * 256 + threadIdx.x;
  long total = (long)Kin * Nout;
  if (idx >= total) return;
  int n = (int)(idx / Kin);
  int k = (int)(idx % Kin);
  dst[idx] = (_Float16)src[(long)k * Nout + n];
}

// context [4,77,768] f32 -> zero-padded [4,96,768] f16
__global__ void ctx_pad(const float* __restrict__ ctx, _Float16* __restrict__ dst) {
  long idx = (long)blockIdx.x * 256 + threadIdx.x;
  long total = (long)NB * TPADC * CTXD;
  if (idx >= total) return;
  int c = (int)(idx % CTXD);
  int r = (int)((idx / CTXD) % TPADC);
  int b = (int)(idx / ((long)CTXD * TPADC));
  float v = (r < TCTX) ? ctx[((long)b * TCTX + r) * CTXD + c] : 0.0f;
  dst[idx] = (_Float16)v;
}

// V [B*Trows, 1024] f16 -> Vt [(B*H)][DHEAD][Trows] f16 (per-head transpose)
__global__ void v_transpose(const _Float16* __restrict__ v, _Float16* __restrict__ vt,
                            int Trows) {
  long idx = (long)blockIdx.x * 256 + threadIdx.x;
  long total = (long)NB * HEADS * DHEAD * Trows;
  if (idx >= total) return;
  int t  = (int)(idx % Trows);
  int d  = (int)((idx / Trows) % DHEAD);
  int hh = (int)((idx / ((long)Trows * DHEAD)) % HEADS);
  int b  = (int)(idx / ((long)Trows * DHEAD * HEADS));
  vt[idx] = v[((long)b * Trows + t) * DIMD + hh * DHEAD + d];
}

// LayerNorm over last dim (1024) of fp32 residual -> f16 normed output
__global__ void layernorm_f16(const float* __restrict__ X, const float* __restrict__ g,
                              const float* __restrict__ bb, _Float16* __restrict__ Y) {
  const int row = blockIdx.x;      // one row per block
  const int tid = threadIdx.x;     // 256 threads, 4 floats each
  const float4 xv = reinterpret_cast<const float4*>(X + (size_t)row * DIMD)[tid];
  float s = xv.x + xv.y + xv.z + xv.w;
  float q = xv.x * xv.x + xv.y * xv.y + xv.z * xv.z + xv.w * xv.w;
  #pragma unroll
  for (int off = 16; off > 0; off >>= 1) {
    s += __shfl_xor(s, off, 32);
    q += __shfl_xor(q, off, 32);
  }
  __shared__ float ss[8], sq[8];
  if ((tid & 31) == 0) { ss[tid >> 5] = s; sq[tid >> 5] = q; }
  __syncthreads();
  float st = 0.f, qt = 0.f;
  #pragma unroll
  for (int i = 0; i < 8; i++) { st += ss[i]; qt += sq[i]; }
  const float mean = st * (1.0f / DIMD);
  const float var  = qt * (1.0f / DIMD) - mean * mean;
  const float rs   = rsqrtf(var + 1e-5f);
  const float4 gv = reinterpret_cast<const float4*>(g)[tid];
  const float4 bv = reinterpret_cast<const float4*>(bb)[tid];
  _Float16* yp = Y + (size_t)row * DIMD + tid * 4;
  yp[0] = (_Float16)((xv.x - mean) * rs * gv.x + bv.x);
  yp[1] = (_Float16)((xv.y - mean) * rs * gv.y + bv.y);
  yp[2] = (_Float16)((xv.z - mean) * rs * gv.z + bv.z);
  yp[3] = (_Float16)((xv.w - mean) * rs * gv.w + bv.w);
}

// ---------------- WMMA GEMM, LDS-staged B with async double-buffering -------
// C = A[M,K] @ Bt[N,K]^T   (Bt = transposed weight, row-major [N,K], f16)
// MODE 0: Cout f16 = result
// MODE 1: Cres f32 += result + bias[n]
// MODE 2: Cout f16 = (res+bias[n]) * gelu(res2+bias2[n])  (Bt2 = gate weights)
// block = 128 threads (4 waves). Block tile = 64(M) x 64(N); wave w owns rows
// m0 = by*64 + w*16; all waves share the N tile. B (and gate B) staged in LDS,
// double-buffered with async global->LDS loads; A fragments stream from global.
// grid = ( N/64 , M/64 )
template <int MODE>
__global__ void gemm_wmma(const _Float16* __restrict__ A,
                          const _Float16* __restrict__ Bt,
                          const _Float16* __restrict__ Bt2,
                          const float* __restrict__ bias,
                          const float* __restrict__ bias2,
                          _Float16* __restrict__ Cout,
                          float* __restrict__ Cres,
                          int M, int N, int K) {
  const int tid  = threadIdx.x;
  const int lane = tid & 31;
  const int wave = tid >> 5;
  const int hi = lane >> 4;       // half-wave select for K packing
  const int ln = lane & 15;
  const int n0 = blockIdx.x * 64;
  const int m0 = blockIdx.y * 64 + wave * 16;
  (void)M;

  constexpr int BROWS  = (MODE == 2) ? 128 : 64;   // lin rows [+ gate rows]
  constexpr int NISSUE = (MODE == 2) ? 4 : 2;      // async issues per wave/step
  __shared__ __align__(16) _Float16 bsh[2][BROWS][32];

  // stage one 32-wide K slice of B (and gate B) into LDS buffer `buf`
  auto stage = [&](int buf, int k0) {
    #pragma unroll
    for (int i = 0; i < NISSUE; i++) {
      const int c   = tid + i * 128;      // 16B chunk id
      const int row = c >> 2;             // 0..BROWS-1
      const int qd  = (c & 3) * 8;        // element offset in K slice
      const _Float16* gsrc =
          (MODE == 2 && row >= 64)
              ? Bt2 + (size_t)(n0 + row - 64) * K + k0 + qd
              : Bt  + (size_t)(n0 + row) * K + k0 + qd;
      copy16_to_lds(gsrc, &bsh[buf][row][qd]);
    }
  };

  v8f acc[4];
  v8f accg[4];
  #pragma unroll
  for (int nt = 0; nt < 4; nt++)
    #pragma unroll
    for (int r = 0; r < 8; r++) { acc[nt][r] = 0.0f; if (MODE == 2) accg[nt][r] = 0.0f; }

  const _Float16* arow = A + (size_t)(m0 + ln) * K + hi * 8;
  const int KSTEPS = K >> 5;

  stage(0, 0);
  for (int it = 0; it < KSTEPS; it++) {
    const int k0  = it << 5;
    const int cur = it & 1;
    if (it + 1 < KSTEPS) {
      stage(cur ^ 1, k0 + 32);            // prefetch next slice into other buffer
      async_wait<NISSUE>();               // current slice's batch has landed
      __builtin_prefetch(arow + k0 + 64, 0, 0);  // global_prefetch A stream
    } else {
      async_wait<0>();
    }
    __syncthreads();                      // all waves' staging visible

    v16h af = frag_load(arow + k0);
    #pragma unroll
    for (int nt = 0; nt < 4; nt++) {
      v16h bf = frag_load(&bsh[cur][nt * 16 + ln][hi * 8]);
      acc[nt] = wmma16(af, bf, acc[nt]);
      if (MODE == 2) {
        v16h bg = frag_load(&bsh[cur][64 + nt * 16 + ln][hi * 8]);
        accg[nt] = wmma16(af, bg, accg[nt]);
      }
    }
    __syncthreads();                      // reads done before buffer reuse
  }

  #pragma unroll
  for (int nt = 0; nt < 4; nt++) {
    #pragma unroll
    for (int r = 0; r < 8; r++) {
      const int m = m0 + r + 8 * hi;      // C layout: lane holds rows r+8*(lane>>4)
      const int n = n0 + nt * 16 + ln;
      const size_t idx = (size_t)m * N + n;
      if (MODE == 0) {
        Cout[idx] = (_Float16)acc[nt][r];
      } else if (MODE == 1) {
        Cres[idx] = Cres[idx] + acc[nt][r] + bias[n];
      } else {
        const float lin  = acc[nt][r]  + bias[n];
        const float gate = accg[nt][r] + bias2[n];
        Cout[idx] = (_Float16)(lin * gelu_tanh(gate));
      }
    }
  }
}

// ---------------- flash attention ----------------
// Q   : [NTOK, 1024] f16   (row = b*SEQ + s)
// Kb  : [NB*Trows, 1024] f16
// Vt  : [(NB*HEADS)][DHEAD][Trows] f16  (per-head transposed V)
// O   : [NTOK, 1024] f16  (concat-head layout)
// block = 128 (4 waves); wave = 16 query rows of one (b,h).
// grid = ( SEQ/64 , NB*HEADS )
__global__ void flash_attn(const _Float16* __restrict__ Q,
                           const _Float16* __restrict__ Kb,
                           const _Float16* __restrict__ Vt,
                           _Float16* __restrict__ O,
                           int Trows, int Tkeys, float scale) {
  const int lane = threadIdx.x & 31;
  const int wave = threadIdx.x >> 5;
  const int hi = lane >> 4;
  const int ln = lane & 15;
  const int bh = blockIdx.y;
  const int b = bh >> 4, h = bh & 15;
  const int qt = blockIdx.x * 4 + wave;   // 16-row query tile

  __shared__ __align__(16) _Float16 psh[4][16][32];   // P exchange, per wave

  // Q fragments for both 32-wide halves of the 64-wide head dim
  const _Float16* qrow = Q + (size_t)(b * SEQ + qt * 16 + ln) * DIMD + h * DHEAD + hi * 8;
  const v16h qa0 = frag_load(qrow);
  const v16h qa1 = frag_load(qrow + 32);

  float mrow[8], lrow[8];
  v8f oa[4];
  #pragma unroll
  for (int r = 0; r < 8; r++) { mrow[r] = -1e30f; lrow[r] = 0.0f; }
  #pragma unroll
  for (int nt = 0; nt < 4; nt++)
    #pragma unroll
    for (int r = 0; r < 8; r++) oa[nt][r] = 0.0f;

  const _Float16* kbase = Kb + (size_t)b * Trows * DIMD + h * DHEAD + hi * 8;
  const _Float16* vbase = Vt + (size_t)bh * DHEAD * Trows;

  for (int j0 = 0; j0 < Trows; j0 += 32) {      // 32 keys per step
    // ---- S = Q K^T (two 16x16 tiles) ----
    v8f sa[2];
    #pragma unroll
    for (int t = 0; t < 2; t++) {
      #pragma unroll
      for (int r = 0; r < 8; r++) sa[t][r] = 0.0f;
      const _Float16* krow = kbase + (size_t)(j0 + t * 16 + ln) * DIMD;
      v16h kf0 = frag_load(krow);
      v16h kf1 = frag_load(krow + 32);
      sa[t] = wmma16(qa0, kf0, sa[t]);
      sa[t] = wmma16(qa1, kf1, sa[t]);
    }
    // ---- scale + key mask ----
    #pragma unroll
    for (int t = 0; t < 2; t++) {
      const bool valid = (j0 + t * 16 + ln) < Tkeys;
      #pragma unroll
      for (int r = 0; r < 8; r++)
        sa[t][r] = valid ? sa[t][r] * scale : -1e30f;
    }
    // ---- online softmax: per-row stats via 16-lane reductions ----
    float alpha[8];
    #pragma unroll
    for (int r = 0; r < 8; r++) {
      float lm = fmaxf(sa[0][r], sa[1][r]);
      #pragma unroll
      for (int off = 8; off > 0; off >>= 1) lm = fmaxf(lm, __shfl_xor(lm, off, 32));
      const float mn = fmaxf(mrow[r], lm);
      const float p0 = __expf(sa[0][r] - mn);
      const float p1 = __expf(sa[1][r] - mn);
      const float al = __expf(mrow[r] - mn);
      float rs = p0 + p1;
      #pragma unroll
      for (int off = 8; off > 0; off >>= 1) rs += __shfl_xor(rs, off, 32);
      lrow[r] = lrow[r] * al + rs;
      mrow[r] = mn;
      alpha[r] = al;
      sa[0][r] = p0; sa[1][r] = p1;
    }
    #pragma unroll
    for (int nt = 0; nt < 4; nt++)
      #pragma unroll
      for (int r = 0; r < 8; r++) oa[nt][r] *= alpha[r];

    // ---- convert P from C-layout to A-fragment layout through LDS ----
    #pragma unroll
    for (int t = 0; t < 2; t++)
      #pragma unroll
      for (int r = 0; r < 8; r++)
        psh[wave][r + 8 * hi][t * 16 + ln] = (_Float16)sa[t][r];
    __syncthreads();
    const v16h pf = frag_load(&psh[wave][ln][hi * 8]);

    // ---- O += P V ----
    #pragma unroll
    for (int nt = 0; nt < 4; nt++) {
      const _Float16* vrow = vbase + (size_t)(nt * 16 + ln) * Trows + j0 + hi * 8;
      v16h vf = frag_load(vrow);
      oa[nt] = wmma16(pf, vf, oa[nt]);
    }
    __syncthreads();
  }

  // ---- normalize and store (concat-head layout) ----
  #pragma unroll
  for (int nt = 0; nt < 4; nt++)
    #pragma unroll
    for (int r = 0; r < 8; r++) {
      const int m = qt * 16 + r + 8 * hi;
      const size_t idx = (size_t)(b * SEQ + m) * DIMD + h * DHEAD + nt * 16 + ln;
      O[idx] = (_Float16)(oa[nt][r] / lrow[r]);
    }
}

// ---------------- host orchestration ----------------
extern "C" void kernel_launch(void* const* d_in, const int* in_sizes, int n_in,
                              void* d_out, int out_size, void* d_ws, size_t ws_size,
                              hipStream_t stream) {
  (void)in_sizes; (void)n_in; (void)out_size; (void)ws_size;
  const float* hidden = (const float*)d_in[0];
  const float* ctx    = (const float*)d_in[1];
  const float* wq1 = (const float*)d_in[2];
  const float* wk1 = (const float*)d_in[3];
  const float* wv1 = (const float*)d_in[4];
  const float* wo1 = (const float*)d_in[5];
  const float* bo1 = (const float*)d_in[6];
  const float* wq2 = (const float*)d_in[7];
  const float* wk2 = (const float*)d_in[8];
  const float* wv2 = (const float*)d_in[9];
  const float* wo2 = (const float*)d_in[10];
  const float* bo2 = (const float*)d_in[11];
  const float* wff1 = (const float*)d_in[12];
  const float* bff1 = (const float*)d_in[13];
  const float* wff2 = (const float*)d_in[14];
  const float* bff2 = (const float*)d_in[15];
  const float* ln1g = (const float*)d_in[16];
  const float* ln1b = (const float*)d_in[17];
  const float* ln2g = (const float*)d_in[18];
  const float* ln2b = (const float*)d_in[19];
  const float* ln3g = (const float*)d_in[20];
  const float* ln3b = (const float*)d_in[21];
  float* out = (float*)d_out;              // fp32 residual h lives here

  // -------- workspace carve-up (f16 buffers, 256B aligned) --------
  char* wsp = (char*)d_ws;
  auto alloc = [&](size_t elems) -> _Float16* {
    _Float16* p = (_Float16*)wsp;
    wsp += ((elems * sizeof(_Float16) + 255) / 256) * 256;
    return p;
  };
  _Float16* wq1T  = alloc((size_t)DIMD * DIMD);
  _Float16* wk1T  = alloc((size_t)DIMD * DIMD);
  _Float16* wv1T  = alloc((size_t)DIMD * DIMD);
  _Float16* wo1T  = alloc((size_t)DIMD * DIMD);
  _Float16* wq2T  = alloc((size_t)DIMD * DIMD);
  _Float16* wk2T  = alloc((size_t)DIMD * CTXD);
  _Float16* wv2T  = alloc((size_t)DIMD * CTXD);
  _Float16* wo2T  = alloc((size_t)DIMD * DIMD);
  _Float16* wff1T = alloc((size_t)(2 * FFD) * DIMD);
  _Float16* wff2T = alloc((size_t)DIMD * FFD);
  _Float16* xnorm = alloc((size_t)NTOK * DIMD);
  _Float16* qbuf  = alloc((size_t)NTOK * DIMD);
  _Float16* kbuf  = alloc((size_t)NTOK * DIMD);
  _Float16* vbuf  = alloc((size_t)NTOK * DIMD);
  _Float16* vT    = alloc((size_t)NB * HEADS * DHEAD * SEQ);
  _Float16* attn  = alloc((size_t)NTOK * DIMD);
  _Float16* ctxp  = alloc((size_t)CROWS * CTXD);
  _Float16* kctx  = alloc((size_t)CROWS * DIMD);
  _Float16* vctx  = alloc((size_t)CROWS * DIMD);
  _Float16* vTc   = alloc((size_t)NB * HEADS * DHEAD * TPADC);
  _Float16* gbuf  = alloc((size_t)NTOK * FFD);

  // h = hidden_states
  (void)hipMemcpyAsync(out, hidden, (size_t)NTOK * DIMD * sizeof(float),
                       hipMemcpyDeviceToDevice, stream);

  // -------- weight cast + transpose --------
  auto wtr = [&](const float* src, _Float16* dst, int Kin, int Nout) {
    long total = (long)Kin * Nout;
    wcast_tr<<<dim3((unsigned)((total + 255) / 256)), dim3(256), 0, stream>>>(src, dst, Kin, Nout);
  };
  wtr(wq1, wq1T, DIMD, DIMD);
  wtr(wk1, wk1T, DIMD, DIMD);
  wtr(wv1, wv1T, DIMD, DIMD);
  wtr(wo1, wo1T, DIMD, DIMD);
  wtr(wq2, wq2T, DIMD, DIMD);
  wtr(wk2, wk2T, CTXD, DIMD);
  wtr(wv2, wv2T, CTXD, DIMD);
  wtr(wo2, wo2T, DIMD, DIMD);
  wtr(wff1, wff1T, DIMD, 2 * FFD);
  wtr(wff2, wff2T, FFD, DIMD);

  const dim3 blk128(128);
  const dim3 gN1024(DIMD / 64, NTOK / 64);         // (16, 128)
  const dim3 gCtxKV(DIMD / 64, CROWS / 64);        // (16, 6)
  const dim3 gGeglu(FFD / 64, NTOK / 64);          // (64, 128)
  const dim3 gFlash(SEQ / 64, NB * HEADS);         // (32, 64)

  // ================= self-attention =================
  layernorm_f16<<<NTOK, 256, 0, stream>>>(out, ln1g, ln1b, xnorm);
  gemm_wmma<0><<<gN1024, blk128, 0, stream>>>(xnorm, wq1T, nullptr, nullptr, nullptr,
                                              qbuf, nullptr, NTOK, DIMD, DIMD);
  gemm_wmma<0><<<gN1024, blk128, 0, stream>>>(xnorm, wk1T, nullptr, nullptr, nullptr,
                                              kbuf, nullptr, NTOK, DIMD, DIMD);
  gemm_wmma<0><<<gN1024, blk128, 0, stream>>>(xnorm, wv1T, nullptr, nullptr, nullptr,
                                              vbuf, nullptr, NTOK, DIMD, DIMD);
  {
    long total = (long)NB * HEADS * DHEAD * SEQ;
    v_transpose<<<dim3((unsigned)((total + 255) / 256)), 256, 0, stream>>>(vbuf, vT, SEQ);
  }
  flash_attn<<<gFlash, blk128, 0, stream>>>(qbuf, kbuf, vT, attn, SEQ, SEQ, ATT_SCALE);
  gemm_wmma<1><<<gN1024, blk128, 0, stream>>>(attn, wo1T, nullptr, bo1, nullptr,
                                              nullptr, out, NTOK, DIMD, DIMD);

  // ================= cross-attention =================
  layernorm_f16<<<NTOK, 256, 0, stream>>>(out, ln2g, ln2b, xnorm);
  gemm_wmma<0><<<gN1024, blk128, 0, stream>>>(xnorm, wq2T, nullptr, nullptr, nullptr,
                                              qbuf, nullptr, NTOK, DIMD, DIMD);
  {
    long total = (long)NB * TPADC * CTXD;
    ctx_pad<<<dim3((unsigned)((total + 255) / 256)), 256, 0, stream>>>(ctx, ctxp);
  }
  gemm_wmma<0><<<gCtxKV, blk128, 0, stream>>>(ctxp, wk2T, nullptr, nullptr, nullptr,
                                              kctx, nullptr, CROWS, DIMD, CTXD);
  gemm_wmma<0><<<gCtxKV, blk128, 0, stream>>>(ctxp, wv2T, nullptr, nullptr, nullptr,
                                              vctx, nullptr, CROWS, DIMD, CTXD);
  {
    long total = (long)NB * HEADS * DHEAD * TPADC;
    v_transpose<<<dim3((unsigned)((total + 255) / 256)), 256, 0, stream>>>(vctx, vTc, TPADC);
  }
  flash_attn<<<gFlash, blk128, 0, stream>>>(qbuf, kctx, vTc, attn, TPADC, TCTX, ATT_SCALE);
  gemm_wmma<1><<<gN1024, blk128, 0, stream>>>(attn, wo2T, nullptr, bo2, nullptr,
                                              nullptr, out, NTOK, DIMD, DIMD);

  // ================= GEGLU MLP =================
  layernorm_f16<<<NTOK, 256, 0, stream>>>(out, ln3g, ln3b, xnorm);
  gemm_wmma<2><<<gGeglu, blk128, 0, stream>>>(xnorm, wff1T, wff1T + (size_t)FFD * DIMD,
                                              bff1, bff1 + FFD, gbuf, nullptr,
                                              NTOK, FFD, DIMD);
  gemm_wmma<1><<<gN1024, blk128, 0, stream>>>(gbuf, wff2T, nullptr, bff2, nullptr,
                                              nullptr, out, NTOK, DIMD, FFD);
}
